// Router_42614665511160
// MI455X (gfx1250) — matrix operand order repaired and checked by reading, not compiled, not run on hardware
//
#include <hip/hip_runtime.h>
#include <hip/hip_bf16.h>

// MoE router: logits = x @ gate_w^T (bf16 WMMA), sigmoid, +bias, top-8,
// gather + normalize. T=16384, DIM=4096, E=64, K=8.
//
// Roofline: x stream = 128 MB is the only mandatory HBM traffic
// (~5.6us @ 23.3 TB/s). gate_w chunks are staged into LDS with async
// global->LDS copies (ASYNCcnt), TRIPLE-buffered and staged two chunks
// ahead (s_wait_asynccnt 0x2, in-order completion) so each 4 KB copy gets
// two full chunk-compute windows to land. A fragments are register
// double-buffered and pinned early in each half via sched_barrier(0).
// B fragments are read with ds_load_b128 (shared-array indexing keeps
// addrspace(3) inference; never fabricate LDS pointers from integers).

#define DIM_C 4096
#define NEXP 64
#define TOPK 8
#define KCHUNK 32
#define NCHUNK (DIM_C / KCHUNK)            // 128 (even)
#define WAVES_PER_BLK 4
#define TOK_PER_WAVE 16                    // one 16-row A tile per wave
#define TOK_PER_BLK (WAVES_PER_BLK * TOK_PER_WAVE)   // 64
#define BLK_THREADS (WAVES_PER_BLK * 32)   // 128
#define BSTRIDE 40                         // bf16 elems per expert row in LDS
                                           // (80 B: 16B-aligned, conflict-free)
#define STAGE_OPS 2                        // async instrs per wave per stage

typedef __bf16 bf16x8  __attribute__((ext_vector_type(8)));
typedef __bf16 bf16x16 __attribute__((ext_vector_type(16)));
typedef float  f32x8   __attribute__((ext_vector_type(8)));

__device__ __forceinline__ bf16x16 join16(bf16x8 lo, bf16x8 hi) {
    return __builtin_shufflevector(lo, hi,
        0,1,2,3,4,5,6,7,8,9,10,11,12,13,14,15);
}

// A-fragment (16-bit A 16x32 ISA layout): per-lane two contiguous 16B runs.
__device__ __forceinline__ bf16x16 load_afrag(const __bf16* p) {
    bf16x8 lo = *reinterpret_cast<const bf16x8*>(p);
    bf16x8 hi = *reinterpret_cast<const bf16x8*>(p + 16);
    return join16(lo, hi);
}

__global__ __launch_bounds__(BLK_THREADS) void router_wmma_kernel(
    const __bf16* __restrict__ x,      // [T, DIM] row-major
    const __bf16* __restrict__ gw,     // [E, DIM] row-major
    const float*  __restrict__ bias,   // [E]
    __hip_bfloat16* __restrict__ out_w,// [T, K] bf16
    int* __restrict__ out_i)           // [T, K] int32
{
    __shared__ __align__(16) __bf16 s_b[3][NEXP * BSTRIDE]; // 3 x 5 KB chunks
    __shared__ float s_scores[TOK_PER_BLK * NEXP];          // 16 KB
    __shared__ float s_bias[NEXP];

    const int tid  = threadIdx.x;
    const int lane = tid & 31;
    const int wv   = tid >> 5;
    const int ln   = lane & 15;   // position within half-wave
    const int hi   = lane >> 4;   // half-wave select

    if (tid < NEXP) s_bias[tid] = bias[tid];

    const int tok_base = blockIdx.x * TOK_PER_BLK + wv * TOK_PER_WAVE;

    // A rows (16-bit A 16x32 ISA layout):
    //   lanes 0-15 : row = ln, K offsets {0..7} and {16..23}
    //   lanes 16-31: row = ln, K offsets {8..15} and {24..31}
    const __bf16* arow = x + (size_t)(tok_base + ln) * DIM_C + (hi ? 8 : 0);

    // ---- async staging of a 64x32 gate_w chunk into LDS --------------------
    // Chunk layout: expert-major, 80 B stride (64 B payload + 16 B pad).
    // 256 granules of 16 B; 128 threads x 2 granules -> 2 async instrs/wave.
    // LBASE is a 32-bit LDS byte offset (low 32 bits of a __shared__ address).
    #define STAGE_CHUNK(BUFIDX, KC)                                            \
        {                                                                      \
            const uint32_t lbase_ =                                            \
                (uint32_t)(uintptr_t)&s_b[(BUFIDX)][0];                        \
            _Pragma("unroll")                                                  \
            for (int j = 0; j < STAGE_OPS; ++j) {                              \
                const int g    = tid + BLK_THREADS * j;  /* 0..255 */          \
                const int e    = g >> 2;                                       \
                const int part = g & 3;                                        \
                const __bf16* src = gw + (size_t)e * DIM_C + (KC) + part * 8;  \
                const uint32_t dst = lbase_ + (uint32_t)(e * (BSTRIDE * 2)     \
                                                         + part * 16);         \
                asm volatile("global_load_async_to_lds_b128 %0, %1, off"      \
                             :: "v"(dst), "v"(src) : "memory");                \
            }                                                                  \
        }

    // 4 expert-group B fragments (ds_load) + 4 WMMAs for one token tile.
    #define CONSUME_CHUNK(BUFIDX, AFRAG)                                       \
        {                                                                      \
            _Pragma("unroll")                                                  \
            for (int grp = 0; grp < 4; ++grp) {                                \
                const __bf16* bp = &s_b[(BUFIDX)]                              \
                                       [(grp * 16 + ln) * BSTRIDE + hi * 8];   \
                bf16x8 blo = *reinterpret_cast<const bf16x8*>(bp);             \
                bf16x8 bhi = *reinterpret_cast<const bf16x8*>(bp + 16);        \
                bf16x16 bfr = join16(blo, bhi);                                \
                if      (grp == 0) acc0 = __builtin_amdgcn_wmma_f32_16x16x32_bf16(false, (AFRAG), false, bfr, (short)0, acc0, false, false); \
                else if (grp == 1) acc1 = __builtin_amdgcn_wmma_f32_16x16x32_bf16(false, (AFRAG), false, bfr, (short)0, acc1, false, false); \
                else if (grp == 2) acc2 = __builtin_amdgcn_wmma_f32_16x16x32_bf16(false, (AFRAG), false, bfr, (short)0, acc2, false, false); \
                else               acc3 = __builtin_amdgcn_wmma_f32_16x16x32_bf16(false, (AFRAG), false, bfr, (short)0, acc3, false, false); \
            }                                                                  \
        }

    // End-of-half sync. DEPTH2: two stages may be outstanding; in-order
    // completion means asynccnt<=2 guarantees the older stage has landed.
    #define PIPE_SYNC_DEPTH2()                                                 \
        __builtin_amdgcn_sched_barrier(0);                                     \
        asm volatile("s_wait_asynccnt 0x2" ::: "memory");                      \
        __syncthreads();                                                       \
        __builtin_amdgcn_sched_barrier(0);
    #define PIPE_SYNC_DRAIN()                                                  \
        __builtin_amdgcn_sched_barrier(0);                                     \
        asm volatile("s_wait_asynccnt 0x0" ::: "memory");                      \
        __syncthreads();                                                       \
        __builtin_amdgcn_sched_barrier(0);

    #define ROT3(V) (V) = ((V) == 2) ? 0 : (V) + 1

    f32x8 acc0 = {}, acc1 = {}, acc2 = {}, acc3 = {};

    int bc = 0;   // buffer to consume this half   (= h % 3)
    int bt = 2;   // buffer to stage into this half (= (h + 2) % 3)

    // Prologue: stage chunks 0 and 1; load chunk-0 A while copies run.
    STAGE_CHUNK(0, 0);
    STAGE_CHUNK(1, KCHUNK);
    bf16x16 a_c = load_afrag(arow);
    PIPE_SYNC_DEPTH2();   // chunk 0 landed (chunk 1 may still be in flight)

    for (int h = 0; h < NCHUNK; h += 2) {
        // ==== even half: consume chunk h ===================================
        {
            if (h + 2 < NCHUNK) STAGE_CHUNK(bt, (h + 2) * KCHUNK);
            const int ka = (h + 1 < NCHUNK ? h + 1 : h) * KCHUNK;
            bf16x16 a_n = load_afrag(arow + ka);        // A double buffer
            const int kpf = h * KCHUNK + 512;
            if (kpf < DIM_C) __builtin_prefetch(arow + kpf, 0, 1);
            __builtin_amdgcn_sched_barrier(0);          // pin loads early

            CONSUME_CHUNK(bc, a_c);
            if (h + 2 < NCHUNK) { PIPE_SYNC_DEPTH2(); } else { PIPE_SYNC_DRAIN(); }

            ROT3(bc); ROT3(bt);
            a_c = a_n;
        }
        // ==== odd half: consume chunk h+1 ==================================
        {
            if (h + 3 < NCHUNK) STAGE_CHUNK(bt, (h + 3) * KCHUNK);
            const int ka = (h + 2 < NCHUNK ? h + 2 : h + 1) * KCHUNK;
            bf16x16 a_n = load_afrag(arow + ka);
            __builtin_amdgcn_sched_barrier(0);          // pin loads early

            CONSUME_CHUNK(bc, a_c);
            if (h + 3 < NCHUNK) { PIPE_SYNC_DEPTH2(); } else { PIPE_SYNC_DRAIN(); }

            ROT3(bc); ROT3(bt);
            a_c = a_n;
        }
    }

    // C/D 16x16 f32 layout: VGPR v, lanes 0-15 -> M=v, N=lane;
    //                               lanes 16-31 -> M=v+8, N=lane-16.
    #pragma unroll
    for (int v = 0; v < 8; ++v) {
        const int t0 = wv * TOK_PER_WAVE + v + hi * 8;
        s_scores[t0 * NEXP +  0 + ln] = acc0[v];
        s_scores[t0 * NEXP + 16 + ln] = acc1[v];
        s_scores[t0 * NEXP + 32 + ln] = acc2[v];
        s_scores[t0 * NEXP + 48 + ln] = acc3[v];
    }
    __syncthreads();

    // sigmoid in place
    for (int i = tid; i < TOK_PER_BLK * NEXP; i += BLK_THREADS) {
        float l = s_scores[i];
        s_scores[i] = 1.0f / (1.0f + __expf(-l));
    }
    __syncthreads();

    // top-8: one thread per token, register-resident scores
    if (tid < TOK_PER_BLK) {
        float sreg[NEXP];
        #pragma unroll
        for (int e = 0; e < NEXP; ++e)
            sreg[e] = s_scores[tid * NEXP + e];

        unsigned long long taken = 0ull;
        float wsel[TOPK];
        int   isel[TOPK];
        float wsum = 0.0f;

        #pragma unroll
        for (int k = 0; k < TOPK; ++k) {
            float best = -3.402823466e38f;
            float bsc  = 0.0f;
            int   bi   = 0;
            #pragma unroll
            for (int e = 0; e < NEXP; ++e) {
                float v = sreg[e] + s_bias[e];
                bool avail = ((taken >> e) & 1ull) == 0ull;
                if (avail && v > best) { best = v; bsc = sreg[e]; bi = e; }
            }
            taken |= (1ull << bi);
            wsel[k] = bsc;
            isel[k] = bi;
            wsum += bsc;
        }

        const float inv = 1.0f / fmaxf(wsum, 1e-12f);
        const size_t t = (size_t)blockIdx.x * TOK_PER_BLK + tid;
        #pragma unroll
        for (int k = 0; k < TOPK; ++k) {
            out_w[t * TOPK + k] = __float2bfloat16(wsel[k] * inv);
            out_i[t * TOPK + k] = isel[k];
        }
    }
}

extern "C" void kernel_launch(void* const* d_in, const int* in_sizes, int n_in,
                              void* d_out, int out_size, void* d_ws, size_t ws_size,
                              hipStream_t stream) {
    (void)n_in; (void)out_size; (void)d_ws; (void)ws_size;

    const __bf16* x    = reinterpret_cast<const __bf16*>(d_in[0]);
    const __bf16* gw   = reinterpret_cast<const __bf16*>(d_in[1]);
    const float*  bias = reinterpret_cast<const float*>(d_in[2]);
    // d_in[3] = topk scalar (fixed to 8 at compile time)

    const int tokens = in_sizes[0] / DIM_C;

    __hip_bfloat16* out_w = reinterpret_cast<__hip_bfloat16*>(d_out);
    int* out_i = reinterpret_cast<int*>(
        reinterpret_cast<char*>(d_out) + (size_t)tokens * TOPK * sizeof(__hip_bfloat16));

    dim3 grid(tokens / TOK_PER_BLK);   // 256 blocks
    dim3 block(BLK_THREADS);           // 128 threads = 4 wave32
    router_wmma_kernel<<<grid, block, 0, stream>>>(x, gw, bias, out_w, out_i);
}
